// ContourProposalNetwork_34059090657772
// MI455X (gfx1250) — matrix operand order repaired and checked by reading, not compiled, not run on hardware
//
#include <hip/hip_runtime.h>
#include <stdint.h>

// ---------------------------------------------------------------------------
// CDNA5 (gfx1250) types for WMMA bf16
// ---------------------------------------------------------------------------
typedef __attribute__((ext_vector_type(16))) __bf16 v16bf;
typedef __attribute__((ext_vector_type(8)))  __bf16 v8bf;
typedef __attribute__((ext_vector_type(8)))  float  v8f;

__device__ __forceinline__ unsigned short f2bf(float f) {
    union { float f; uint32_t u; } c; c.f = f;
    uint32_t u = c.u;
    uint32_t r = u + 0x7FFFu + ((u >> 16) & 1u);   // round-to-nearest-even
    return (unsigned short)(r >> 16);
}
__device__ __forceinline__ float bf2f(unsigned short h) {
    union { uint32_t u; float f; } c; c.u = ((uint32_t)h) << 16;
    return c.f;
}

// ---------------------------------------------------------------------------
// Zero-fill (padding borders must be 0 every call; fill whole activation pool)
// ---------------------------------------------------------------------------
__global__ void fill_zero_u32(uint32_t* p, size_t n) {
    size_t i  = (size_t)blockIdx.x * blockDim.x + threadIdx.x;
    size_t st = (size_t)gridDim.x * blockDim.x;
    for (; i < n; i += st) p[i] = 0u;
}

// ---------------------------------------------------------------------------
// Weights: f32 OIHW -> bf16 [oc][ky*3+kx][ic]  (K-runs over ic contiguous)
// ---------------------------------------------------------------------------
__global__ void pack_w3x3(const float* __restrict__ w, unsigned short* __restrict__ out,
                          int Cout, int Cin) {
    size_t n  = (size_t)Cout * Cin * 9;
    size_t i  = (size_t)blockIdx.x * blockDim.x + threadIdx.x;
    size_t st = (size_t)gridDim.x * blockDim.x;
    for (; i < n; i += st) {
        int ic   = (int)(i % Cin);
        size_t t = i / Cin;
        int kidx = (int)(t % 9);
        int oc   = (int)(t / 9);
        out[i] = f2bf(w[((size_t)oc * Cin + ic) * 9 + kidx]);
    }
}

// ---------------------------------------------------------------------------
// Features: f32 NCHW -> bf16 padded HWC  [(H+2)][(W+2)][C], interior only
// ---------------------------------------------------------------------------
__global__ void to_hwc_pad(const float* __restrict__ x, unsigned short* __restrict__ xp,
                           int C, int H, int W) {
    size_t n  = (size_t)C * H * W;
    int P = W + 2;
    size_t i  = (size_t)blockIdx.x * blockDim.x + threadIdx.x;
    size_t st = (size_t)gridDim.x * blockDim.x;
    for (; i < n; i += st) {
        int c    = (int)(i % C);
        size_t t = i / C;
        int xx   = (int)(t % W);
        int yy   = (int)(t / W);
        xp[(((size_t)(yy + 1)) * P + (xx + 1)) * C + c] =
            f2bf(x[((size_t)c * H + yy) * W + xx]);
    }
}

// ---------------------------------------------------------------------------
// Implicit-GEMM 3x3 SAME conv, bf16 WMMA, fused bias+ReLU, bf16 padded-HWC out.
// Block: 128 threads = 4 waves. Block tile: 128 oc x 32 pixels (one row seg).
// Wave tile:  32 oc x 32 pixels = 4 accumulators, 4 WMMAs per K-step on
//             2 A-fragments + 2 B-fragments (2 b128 loads per WMMA).
// ---------------------------------------------------------------------------
__global__ __launch_bounds__(128) void conv3x3_wmma(
    const unsigned short* __restrict__ Xp,   // padded HWC bf16 [(H+2)(W+2)][Cin]
    const unsigned short* __restrict__ Wp,   // packed bf16 [oc][9][Cin]
    const float* __restrict__ bias,
    unsigned short* __restrict__ Yp,         // padded HWC bf16 [(H+2)(W+2)][Cout]
    int H, int W, int Cin, int Cout)
{
    const int lane   = threadIdx.x & 31;
    const int wv     = threadIdx.x >> 5;            // wave 0..3
    const int tilesx = W >> 5;
    const int x0     = (blockIdx.x % tilesx) << 5;  // 32-pixel row segment
    const int y      = blockIdx.x / tilesx;
    const int oc0    = (blockIdx.y << 7) + (wv << 5);  // 32 oc per wave
    const int P      = W + 2;
    const int col    = lane & 15;
    const int hi     = lane >> 4;                   // half-wave select
    const int aoff   = hi ? 16 : 0;                 // A: K 0..15 vs 16..31
    const int boff   = hi ? 16 : 0;                 // B: K 0..15 vs 16..31

    v8f acc00 = {}; v8f acc01 = {};                 // oc[0..15]  x px[0..15 / 16..31]
    v8f acc10 = {}; v8f acc11 = {};                 // oc[16..31] x px[0..15 / 16..31]

    const unsigned short* wrowA = Wp + (size_t)(oc0 + col)      * 9 * Cin;
    const unsigned short* wrowB = Wp + (size_t)(oc0 + 16 + col) * 9 * Cin;

    for (int kidx = 0; kidx < 9; ++kidx) {
        const int dy = kidx / 3, dx = kidx % 3;
        const unsigned short* xrow = Xp + ((size_t)(y + dy) * P + (x0 + dx)) * Cin;
        const unsigned short* wr0  = wrowA + (size_t)kidx * Cin;
        const unsigned short* wr1  = wrowB + (size_t)kidx * Cin;
        for (int icb = 0; icb < Cin; icb += 32) {
            // A fragments: 16x32 bf16 weights (contiguous K: lane half h holds K 16h..16h+15)
            v8bf a0lo = *(const v8bf*)(wr0 + icb + aoff);
            v8bf a0hi = *(const v8bf*)(wr0 + icb + 8 + aoff);
            v8bf a1lo = *(const v8bf*)(wr1 + icb + aoff);
            v8bf a1hi = *(const v8bf*)(wr1 + icb + 8 + aoff);
            v16bf a0 = __builtin_shufflevector(a0lo, a0hi,
                0, 1, 2, 3, 4, 5, 6, 7, 8, 9, 10, 11, 12, 13, 14, 15);
            v16bf a1 = __builtin_shufflevector(a1lo, a1hi,
                0, 1, 2, 3, 4, 5, 6, 7, 8, 9, 10, 11, 12, 13, 14, 15);
            // B fragments: 32x16 bf16, one contiguous 32B channel run per lane
            v16bf b0 = *(const v16bf*)(xrow + ((size_t)col       ) * Cin + icb + boff);
            v16bf b1 = *(const v16bf*)(xrow + ((size_t)(col + 16)) * Cin + icb + boff);

            acc00 = __builtin_amdgcn_wmma_f32_16x16x32_bf16(
                        false, a0, false, b0, (short)0, acc00, false, false);
            acc01 = __builtin_amdgcn_wmma_f32_16x16x32_bf16(
                        false, a0, false, b1, (short)0, acc01, false, false);
            acc10 = __builtin_amdgcn_wmma_f32_16x16x32_bf16(
                        false, a1, false, b0, (short)0, acc10, false, false);
            acc11 = __builtin_amdgcn_wmma_f32_16x16x32_bf16(
                        false, a1, false, b1, (short)0, acc11, false, false);
        }
        // keep next tap's activation row warm in L2 / WGP$
        if (kidx < 8) {
            const int ndy = (kidx + 1) / 3, ndx = (kidx + 1) % 3;
            __builtin_prefetch(Xp + ((size_t)(y + ndy) * P + (x0 + ndx)) * Cin, 0, 1);
        }
    }

    // Epilogue: D lane layout => 8 consecutive oc per lane -> 16B packed stores.
    #pragma unroll
    for (int r = 0; r < 2; ++r) {                   // oc sub-tile (0..15 / 16..31)
        const int mb = oc0 + (r << 4) + (hi ? 8 : 0);
        float b8[8];
        #pragma unroll
        for (int v = 0; v < 8; ++v) b8[v] = bias[mb + v];

        #pragma unroll
        for (int t = 0; t < 2; ++t) {               // pixel sub-tile
            v8f acc = r ? (t ? acc11 : acc10) : (t ? acc01 : acc00);
            const int n = col + (t << 4);
            uint32_t pk[4];
            #pragma unroll
            for (int q = 0; q < 4; ++q) {
                float f0 = acc[2 * q + 0] + b8[2 * q + 0];
                float f1 = acc[2 * q + 1] + b8[2 * q + 1];
                f0 = f0 > 0.f ? f0 : 0.f;
                f1 = f1 > 0.f ? f1 : 0.f;
                pk[q] = (uint32_t)f2bf(f0) | ((uint32_t)f2bf(f1) << 16);
            }
            uint4 out4 = make_uint4(pk[0], pk[1], pk[2], pk[3]);
            *(uint4*)(Yp + (((size_t)(y + 1)) * P + (x0 + n + 1)) * Cout + mb) = out4;
        }
    }
}

// ---------------------------------------------------------------------------
// 1x1 conv (256 -> 1) + sigmoid  => classification (f32)
// ---------------------------------------------------------------------------
__global__ void conv1x1_sig(const unsigned short* __restrict__ X, const float* __restrict__ w,
                            const float* __restrict__ b, float* __restrict__ out,
                            int H, int W, int Cin) {
    int idx = blockIdx.x * blockDim.x + threadIdx.x;
    if (idx >= H * W) return;
    int xx = idx % W, yy = idx / W;
    int P = W + 2;
    const unsigned short* px = X + (((size_t)(yy + 1)) * P + (xx + 1)) * Cin;
    float s = b[0];
    for (int c = 0; c < Cin; ++c) s += bf2f(px[c]) * w[c];
    out[idx] = 1.f / (1.f + __expf(-s));
}

// ---------------------------------------------------------------------------
// 1x1 conv (256 -> 2)  => refinement field f32 [H][W][2]
// ---------------------------------------------------------------------------
__global__ void conv1x1_2ch(const unsigned short* __restrict__ X, const float* __restrict__ w,
                            const float* __restrict__ b, float* __restrict__ field,
                            int H, int W, int Cin) {
    int idx = blockIdx.x * blockDim.x + threadIdx.x;
    if (idx >= H * W) return;
    int xx = idx % W, yy = idx / W;
    int P = W + 2;
    const unsigned short* px = X + (((size_t)(yy + 1)) * P + (xx + 1)) * Cin;
    float s0 = b[0], s1 = b[1];
    for (int c = 0; c < Cin; ++c) {
        float v = bf2f(px[c]);
        s0 += v * w[c];
        s1 += v * w[Cin + c];
    }
    field[(size_t)idx * 2 + 0] = s0;
    field[(size_t)idx * 2 + 1] = s1;
}

// ---------------------------------------------------------------------------
// binary_map = round(bilinear_upsample2_align(classification >= 0.5))
// ---------------------------------------------------------------------------
__global__ void upsample_bin(const float* __restrict__ cls, float* __restrict__ out,
                             int H, int W) {
    int OW = 2 * W, OH = 2 * H;
    int idx = blockIdx.x * blockDim.x + threadIdx.x;
    if (idx >= OH * OW) return;
    int j = idx % OW, i = idx / OW;
    float sy = (float)i * (float)(H - 1) / (float)(OH - 1);
    float sx = (float)j * (float)(W - 1) / (float)(OW - 1);
    int y0 = (int)floorf(sy); int y1 = min(y0 + 1, H - 1);
    int x0 = (int)floorf(sx); int x1 = min(x0 + 1, W - 1);
    float wy = sy - (float)y0, wx = sx - (float)x0;
    float v00 = cls[y0 * W + x0] >= 0.5f ? 1.f : 0.f;
    float v01 = cls[y0 * W + x1] >= 0.5f ? 1.f : 0.f;
    float v10 = cls[y1 * W + x0] >= 0.5f ? 1.f : 0.f;
    float v11 = cls[y1 * W + x1] >= 0.5f ? 1.f : 0.f;
    float v = (v00 * (1.f - wx) + v01 * wx) * (1.f - wy)
            + (v10 * (1.f - wx) + v11 * wx) * wy;
    out[idx] = rintf(v);   // half-to-even like jnp.round
}

// ---------------------------------------------------------------------------
// Iterative contour refinement: bilinear sample field, step, clip.
// ---------------------------------------------------------------------------
__global__ void refine_contours(const float* __restrict__ field, const float* __restrict__ cc,
                                const int* __restrict__ niter, float* __restrict__ out,
                                int N, int H, int W) {
    int p = blockIdx.x * blockDim.x + threadIdx.x;
    if (p >= N) return;
    float x = cc[(size_t)p * 2 + 0];
    float y = cc[(size_t)p * 2 + 1];
    const float hix = (float)(W - 1), hiy = (float)(H - 1);
    int it = *niter;
    for (int k = 0; k < it; ++k) {
        float xs = fminf(fmaxf(x, 0.f), hix);
        float ys = fminf(fmaxf(y, 0.f), hiy);
        int x0 = (int)floorf(xs); int x1 = min(x0 + 1, W - 1);
        int y0 = (int)floorf(ys); int y1 = min(y0 + 1, H - 1);
        float wx = xs - (float)x0, wy = ys - (float)y0;
        const float* f00 = field + ((size_t)y0 * W + x0) * 2;
        const float* f01 = field + ((size_t)y0 * W + x1) * 2;
        const float* f10 = field + ((size_t)y1 * W + x0) * 2;
        const float* f11 = field + ((size_t)y1 * W + x1) * 2;
        float ox = (f00[0] * (1.f - wx) + f01[0] * wx) * (1.f - wy)
                 + (f10[0] * (1.f - wx) + f11[0] * wx) * wy;
        float oy = (f00[1] * (1.f - wx) + f01[1] * wx) * (1.f - wy)
                 + (f10[1] * (1.f - wx) + f11[1] * wx) * wy;
        x = fminf(fmaxf(x + ox, 0.f), hix);
        y = fminf(fmaxf(y + oy, 0.f), hiy);
    }
    out[(size_t)p * 2 + 0] = x;
    out[(size_t)p * 2 + 1] = y;
}

// ---------------------------------------------------------------------------
// Host launcher
// ---------------------------------------------------------------------------
extern "C" void kernel_launch(void* const* d_in, const int* in_sizes, int n_in,
                              void* d_out, int out_size, void* d_ws, size_t ws_size,
                              hipStream_t stream) {
    (void)in_sizes; (void)n_in; (void)out_size; (void)ws_size;

    const float* features_1 = (const float*)d_in[0];   // (1,512,256,256)
    const float* features_2 = (const float*)d_in[1];   // (1,512,128,128)
    const float* cc         = (const float*)d_in[2];   // (1,600,64,2)
    const float* cw1 = (const float*)d_in[3];  const float* cb1 = (const float*)d_in[4];
    const float* cw2 = (const float*)d_in[5];  const float* cb2 = (const float*)d_in[6];
    const float* cw3 = (const float*)d_in[7];  const float* cb3 = (const float*)d_in[8];
    const float* cw4 = (const float*)d_in[9];  const float* cb4 = (const float*)d_in[10];
    const float* rw1 = (const float*)d_in[11]; const float* rb1 = (const float*)d_in[12];
    const float* rw2 = (const float*)d_in[13]; const float* rb2 = (const float*)d_in[14];
    const float* rw3 = (const float*)d_in[15]; const float* rb3 = (const float*)d_in[16];
    const float* rw4 = (const float*)d_in[17]; const float* rb4 = (const float*)d_in[18];
    const int*   num_iter = (const int*)d_in[19];

    float* out_cls  = (float*)d_out;              // 128*128
    float* out_bin  = out_cls + 128 * 128;        // 256*256
    float* out_cont = out_bin + 256 * 256;        // 600*64*2

    // ---- workspace layout (256B aligned slices) ----
    char* ws = (char*)d_ws;
    size_t off = 0;
    auto carve = [&](size_t bytes) -> char* {
        char* p = ws + off;
        off += (bytes + 255) & ~(size_t)255;
        return p;
    };
    const size_t P256 = 258 * 258, P128 = 130 * 130;
    unsigned short* f1p = (unsigned short*)carve(P256 * 512  * 2);
    unsigned short* r1p = (unsigned short*)carve(P256 * 1024 * 2);
    unsigned short* r2p = (unsigned short*)carve(P256 * 512  * 2);
    unsigned short* r3p = (unsigned short*)carve(P256 * 256  * 2);
    unsigned short* f2p = (unsigned short*)carve(P128 * 512  * 2);
    unsigned short* c1p = (unsigned short*)carve(P128 * 1024 * 2);
    unsigned short* c2p = (unsigned short*)carve(P128 * 512  * 2);
    unsigned short* c3p = (unsigned short*)carve(P128 * 256  * 2);
    size_t act_bytes = off;                                   // all activations
    unsigned short* cw1p = (unsigned short*)carve((size_t)1024 * 9 * 512  * 2);
    unsigned short* cw2p = (unsigned short*)carve((size_t)512  * 9 * 1024 * 2);
    unsigned short* cw3p = (unsigned short*)carve((size_t)256  * 9 * 512  * 2);
    unsigned short* rw1p = (unsigned short*)carve((size_t)1024 * 9 * 512  * 2);
    unsigned short* rw2p = (unsigned short*)carve((size_t)512  * 9 * 1024 * 2);
    unsigned short* rw3p = (unsigned short*)carve((size_t)256  * 9 * 512  * 2);
    float* rfield = (float*)carve((size_t)256 * 256 * 2 * 4);

    // ---- zero activation pool (deterministic padding borders) ----
    fill_zero_u32<<<2048, 256, 0, stream>>>((uint32_t*)ws, act_bytes / 4);

    // ---- pack weights to bf16 [oc][tap][ic] ----
    pack_w3x3<<<2048, 256, 0, stream>>>(cw1, cw1p, 1024, 512);
    pack_w3x3<<<2048, 256, 0, stream>>>(cw2, cw2p, 512, 1024);
    pack_w3x3<<<1024, 256, 0, stream>>>(cw3, cw3p, 256, 512);
    pack_w3x3<<<2048, 256, 0, stream>>>(rw1, rw1p, 1024, 512);
    pack_w3x3<<<2048, 256, 0, stream>>>(rw2, rw2p, 512, 1024);
    pack_w3x3<<<1024, 256, 0, stream>>>(rw3, rw3p, 256, 512);

    // ---- features -> bf16 padded HWC ----
    to_hwc_pad<<<4096, 256, 0, stream>>>(features_1, f1p, 512, 256, 256);
    to_hwc_pad<<<2048, 256, 0, stream>>>(features_2, f2p, 512, 128, 128);

    // ---- classification branch (128x128) ----
    {
        dim3 blk(128);
        conv3x3_wmma<<<dim3((128 / 32) * 128, 1024 / 128), blk, 0, stream>>>(
            f2p, cw1p, cb1, c1p, 128, 128, 512, 1024);
        conv3x3_wmma<<<dim3((128 / 32) * 128, 512 / 128), blk, 0, stream>>>(
            c1p, cw2p, cb2, c2p, 128, 128, 1024, 512);
        conv3x3_wmma<<<dim3((128 / 32) * 128, 256 / 128), blk, 0, stream>>>(
            c2p, cw3p, cb3, c3p, 128, 128, 512, 256);
        conv1x1_sig<<<(128 * 128 + 255) / 256, 256, 0, stream>>>(
            c3p, cw4, cb4, out_cls, 128, 128, 256);
        upsample_bin<<<(256 * 256 + 255) / 256, 256, 0, stream>>>(
            out_cls, out_bin, 128, 128);
    }

    // ---- refinement branch (256x256) ----
    {
        dim3 blk(128);
        conv3x3_wmma<<<dim3((256 / 32) * 256, 1024 / 128), blk, 0, stream>>>(
            f1p, rw1p, rb1, r1p, 256, 256, 512, 1024);
        conv3x3_wmma<<<dim3((256 / 32) * 256, 512 / 128), blk, 0, stream>>>(
            r1p, rw2p, rb2, r2p, 256, 256, 1024, 512);
        conv3x3_wmma<<<dim3((256 / 32) * 256, 256 / 128), blk, 0, stream>>>(
            r2p, rw3p, rb3, r3p, 256, 256, 512, 256);
        conv1x1_2ch<<<(256 * 256 + 255) / 256, 256, 0, stream>>>(
            r3p, rw4, rb4, rfield, 256, 256, 256);
    }

    // ---- contour refinement ----
    refine_contours<<<(600 * 64 + 255) / 256, 256, 0, stream>>>(
        rfield, cc, num_iter, out_cont, 600 * 64, 256, 256);
}
